// InteractionBlock_63531156242866
// MI455X (gfx1250) — matrix operand order repaired and checked by reading, not compiled, not run on hardware
//
#include <hip/hip_runtime.h>
#include <hip/hip_bf16.h>

typedef __bf16 bf16;
typedef __attribute__((ext_vector_type(16))) __bf16 v16bf;
typedef __attribute__((ext_vector_type(8)))  float  v8f;

#define F    128
#define TPAD 136   // padded LDS tile row stride (bf16 elems): 272B = 68 dwords
                   // -> row r starts at bank (r*68)%64 = 4r%64, killing the
                   //    16-way conflicts a 256B stride would cause on ds_load_b128
#define ADIM 64
#define NRBF 6
#define NSBF 42
#define BDIM 8
#define MATFRAG (4 * 8 * 32 * 16)  // bf16 elements in one packed 128x128 weight

__device__ __forceinline__ float swishf(float x) {
  return x / (1.0f + __expf(-x));
}

__device__ __forceinline__ v8f wmma_bf16(v16bf a, v16bf b, v8f c) {
  // 8 args: (neg_a, A, neg_b, B, c_mod, C, reuse_a, reuse_b)
  return __builtin_amdgcn_wmma_f32_16x16x32_bf16(false, a, false, b, (short)0, c,
                                                 false, false);
}

// A-fragment (16x32 bf16) from a row-major fp32 [16 x ld] tile at K offset k0.
// Lane l<16 holds row l, K = k0+{0..7, 16..23}; lane l>=16 holds row l-16,
// K = k0+{8..15, 24..31}  (ISA 7.12.2, 16-bit A 16x32 layout).
__device__ __forceinline__ v16bf load_a_f32(const float* __restrict__ base, int ld,
                                            int k0, int lane) {
  int r = lane & 15;
  int h = (lane >> 4) << 3;
  const float* p0 = base + (size_t)r * ld + k0 + h;
  const float* p1 = p0 + 16;
  v16bf a;
#pragma unroll
  for (int j = 0; j < 8; ++j) {
    a[j]     = (bf16)p0[j];
    a[j + 8] = (bf16)p1[j];
  }
  return a;
}

__device__ __forceinline__ v16bf load_a_bf16(const bf16* base, int ld, int k0,
                                             int lane) {
  int r = lane & 15;
  int h = (lane >> 4) << 3;
  const bf16* p0 = base + r * ld + k0 + h;
  const bf16* p1 = p0 + 16;
  v16bf a;
#pragma unroll
  for (int j = 0; j < 8; ++j) {
    a[j]     = p0[j];
    a[j + 8] = p1[j];
  }
  return a;
}

// Packed B-fragment: 32 B contiguous per lane.
__device__ __forceinline__ v16bf load_frag(const bf16* __restrict__ P, int NT,
                                           int kc, int nt, int lane) {
  const v16bf* p = (const v16bf*)P;
  return p[((kc * NT + nt) << 5) + lane];
}

// Pack row-major fp32 W[KxN] into bf16 B-fragment layout:
// elem j: k = kc*32 + (lane>>4)*8 + (j&7) + (j>>3)*16, n = nt*16 + (lane&15).
__global__ void k_pack(const float* __restrict__ W, bf16* __restrict__ out, int K,
                       int N) {
  int idx = blockIdx.x * blockDim.x + threadIdx.x;
  int total = (K >> 5) * (N >> 4) * 32;
  if (idx >= total) return;
  int lane = idx & 31;
  int f = idx >> 5;
  int ntiles = N >> 4;
  int nt = f % ntiles, kc = f / ntiles;
  int c = lane & 15;
  int kb = (lane >> 4) << 3;
#pragma unroll
  for (int j = 0; j < 16; ++j) {
    int k = (kc << 5) + kb + (j & 7) + ((j >> 3) << 4);
    int n = (nt << 4) + c;
    out[(size_t)idx * 16 + j] = (bf16)W[(size_t)k * N + n];
  }
}

__global__ void k_zero(float* __restrict__ p, size_t n) {
  size_t i = (size_t)blockIdx.x * blockDim.x + threadIdx.x;
  if (i < n) p[i] = 0.0f;
}

// Write C-layout accumulators (8 N-tiles of v8f) into a row-major bf16 [16][TPAD]
// LDS tile. C layout: VGPR v, lane l -> row = v + 8*(l>=16), col = l&15.
__device__ __forceinline__ void store_tile(const v8f m[8], bf16* tile, int lane) {
  __syncthreads();
  int col = lane & 15;
  int ro = (lane & 16) >> 1;
#pragma unroll
  for (int nt = 0; nt < 8; ++nt)
#pragma unroll
    for (int v = 0; v < 8; ++v)
      tile[(v + ro) * TPAD + (nt << 4) + col] = (bf16)m[nt][v];
  __syncthreads();
}

// m += swish(swish(m@W0+b0)@W1+b1)   (one residual layer, F=128)
__device__ void residual128(v8f m[8], bf16* tile, const bf16* __restrict__ P,
                            const float* __restrict__ b, int lane) {
  int col = lane & 15;
  v16bf a[4];
  v8f h[8];
  store_tile(m, tile, lane);
#pragma unroll
  for (int kc = 0; kc < 4; ++kc) a[kc] = load_a_bf16(tile, TPAD, kc << 5, lane);
#pragma unroll
  for (int nt = 0; nt < 8; ++nt) {
    v8f c = {};
#pragma unroll
    for (int kc = 0; kc < 4; ++kc)
      c = wmma_bf16(a[kc], load_frag(P, 8, kc, nt, lane), c);
    float bb = b[(nt << 4) + col];
#pragma unroll
    for (int v = 0; v < 8; ++v) c[v] = swishf(c[v] + bb);
    h[nt] = c;
  }
  store_tile(h, tile, lane);
#pragma unroll
  for (int kc = 0; kc < 4; ++kc) a[kc] = load_a_bf16(tile, TPAD, kc << 5, lane);
#pragma unroll
  for (int nt = 0; nt < 8; ++nt) {
    v8f c = {};
#pragma unroll
    for (int kc = 0; kc < 4; ++kc)
      c = wmma_bf16(a[kc], load_frag(P + MATFRAG, 8, kc, nt, lane), c);
    float bb = b[F + (nt << 4) + col];
#pragma unroll
    for (int v = 0; v < 8; ++v) m[nt][v] += swishf(c[v] + bb);
  }
}

// Phase 1: m_ang = swish( (swish(X@W_kj+b_kj) * rbf_p) @ W_down ), stored bf16 [E,64]
__global__ __launch_bounds__(256) void k_edge_pre(
    const float* __restrict__ m_input, const float* __restrict__ rbf,
    const float* __restrict__ W_rbf1, const float* __restrict__ W_rbf2,
    const float* __restrict__ b_kj, const bf16* __restrict__ Pkj,
    const bf16* __restrict__ Pdn, bf16* __restrict__ m_ang) {
  __shared__ bf16 tiles[8][16 * TPAD];
  __shared__ float rbf8[8][16][BDIM];
  int lane = threadIdx.x & 31;
  int wave = threadIdx.x >> 5;
  bf16* tile = tiles[wave];
  int e0 = (blockIdx.x * 8 + wave) << 4;
  int col = lane & 15;
  int ro = (lane & 16) >> 1;

  if (lane < 16) {  // rbf @ W_rbf1 for this wave's 16 edges
    const float* rr = rbf + (size_t)(e0 + lane) * NRBF;
    float r6[NRBF];
#pragma unroll
    for (int i = 0; i < NRBF; ++i) r6[i] = rr[i];
#pragma unroll
    for (int b = 0; b < BDIM; ++b) {
      float s = 0.f;
#pragma unroll
      for (int i = 0; i < NRBF; ++i) s += r6[i] * W_rbf1[i * BDIM + b];
      rbf8[wave][lane][b] = s;
    }
  }

  v16bf a[4];
#pragma unroll
  for (int kc = 0; kc < 4; ++kc)
    a[kc] = load_a_f32(m_input + (size_t)e0 * F, F, kc << 5, lane);

  __syncthreads();

#pragma unroll
  for (int nt = 0; nt < 8; ++nt) {
    v8f c = {};
#pragma unroll
    for (int kc = 0; kc < 4; ++kc)
      c = wmma_bf16(a[kc], load_frag(Pkj, 8, kc, nt, lane), c);
    int n = (nt << 4) + col;
    float bb = b_kj[n];
    float w2[BDIM];
#pragma unroll
    for (int b = 0; b < BDIM; ++b) w2[b] = W_rbf2[b * F + n];
#pragma unroll
    for (int v = 0; v < 8; ++v) {
      int r = v + ro;
      float rp = 0.f;
#pragma unroll
      for (int b = 0; b < BDIM; ++b) rp += rbf8[wave][r][b] * w2[b];
      tile[r * TPAD + n] = (bf16)(swishf(c[v] + bb) * rp);
    }
  }
  __syncthreads();

  v16bf a2[4];
#pragma unroll
  for (int kc = 0; kc < 4; ++kc) a2[kc] = load_a_bf16(tile, TPAD, kc << 5, lane);
#pragma unroll
  for (int nt = 0; nt < 4; ++nt) {
    v8f c = {};
#pragma unroll
    for (int kc = 0; kc < 4; ++kc)
      c = wmma_bf16(a2[kc], load_frag(Pdn, 4, kc, nt, lane), c);
#pragma unroll
    for (int v = 0; v < 8; ++v)
      m_ang[(size_t)(e0 + v + ro) * ADIM + (nt << 4) + col] = (bf16)swishf(c[v]);
  }
}

// Phase 2: triplets — sbf projection, gather m_ang, scatter-add fp32 into agg.
__global__ __launch_bounds__(256) void k_triplet(
    const float* __restrict__ sbf, const int* __restrict__ red,
    const int* __restrict__ expk, const float* __restrict__ W_sbf1,
    const float* __restrict__ W_sbf2, const bf16* __restrict__ m_ang,
    float* __restrict__ agg, int T) {
  __shared__ float w1[NSBF * BDIM];
  __shared__ float w2[BDIM * ADIM];
  for (int i = threadIdx.x; i < NSBF * BDIM; i += 256) w1[i] = W_sbf1[i];
  for (int i = threadIdx.x; i < BDIM * ADIM; i += 256) w2[i] = W_sbf2[i];
  __syncthreads();
  int t = blockIdx.x * 256 + threadIdx.x;
  if (t >= T) return;
  const float* sr = sbf + (size_t)t * NSBF;
  float s8[BDIM] = {};
  for (int r = 0; r < NSBF; ++r) {
    float sv = sr[r];
#pragma unroll
    for (int b = 0; b < BDIM; ++b) s8[b] += sv * w1[r * BDIM + b];
  }
  int kj = expk[t], ji = red[t];
  const bf16* mr = m_ang + (size_t)kj * ADIM;
  float* ar = agg + (size_t)ji * ADIM;
#pragma unroll 4
  for (int aI = 0; aI < ADIM; ++aI) {
    float sp = 0.f;
#pragma unroll
    for (int b = 0; b < BDIM; ++b) sp += s8[b] * w2[b * ADIM + aI];
    float val = sp * (float)mr[aI];
    __hip_atomic_fetch_add(&ar[aI], val, __ATOMIC_RELAXED, __HIP_MEMORY_SCOPE_AGENT);
  }
}

// Phase 3: m_ji + prop, residual-before, final + skip, residuals-after, write out.
__global__ __launch_bounds__(256) void k_edge_post(
    const float* __restrict__ m_input, const float* __restrict__ agg,
    const bf16* __restrict__ Pji, const float* __restrict__ b_ji,
    const bf16* __restrict__ Pup, const bf16* __restrict__ Prb,
    const float* __restrict__ brb, const bf16* __restrict__ Pfin,
    const float* __restrict__ bfin, const bf16* __restrict__ Pra,
    const float* __restrict__ bra, float* __restrict__ out) {
  __shared__ bf16 tiles[8][16 * TPAD];
  int lane = threadIdx.x & 31;
  int wave = threadIdx.x >> 5;
  bf16* tile = tiles[wave];
  int e0 = (blockIdx.x * 8 + wave) << 4;
  int col = lane & 15;
  int ro = (lane & 16) >> 1;
  const float* Xb = m_input + (size_t)e0 * F;

  v16bf a[4];
#pragma unroll
  for (int kc = 0; kc < 4; ++kc) a[kc] = load_a_f32(Xb, F, kc << 5, lane);

  v8f m[8];
#pragma unroll
  for (int nt = 0; nt < 8; ++nt) {  // m_ji = swish(X@W_ji + b_ji)
    v8f c = {};
#pragma unroll
    for (int kc = 0; kc < 4; ++kc)
      c = wmma_bf16(a[kc], load_frag(Pji, 8, kc, nt, lane), c);
    float bb = b_ji[(nt << 4) + col];
#pragma unroll
    for (int v = 0; v < 8; ++v) c[v] = swishf(c[v] + bb);
    m[nt] = c;
  }

  v16bf au[2];  // prop = swish(agg @ W_up)
#pragma unroll
  for (int kc = 0; kc < 2; ++kc)
    au[kc] = load_a_f32(agg + (size_t)e0 * ADIM, ADIM, kc << 5, lane);
#pragma unroll
  for (int nt = 0; nt < 8; ++nt) {
    v8f c = {};
#pragma unroll
    for (int kc = 0; kc < 2; ++kc)
      c = wmma_bf16(au[kc], load_frag(Pup, 8, kc, nt, lane), c);
#pragma unroll
    for (int v = 0; v < 8; ++v) m[nt][v] += swishf(c[v]);
  }

  residual128(m, tile, Prb, brb, lane);  // NB = 1

  // m = swish(m @ W_final + b_final) + m_input
  store_tile(m, tile, lane);
#pragma unroll
  for (int kc = 0; kc < 4; ++kc) a[kc] = load_a_bf16(tile, TPAD, kc << 5, lane);
#pragma unroll
  for (int nt = 0; nt < 8; ++nt) {
    v8f c = {};
#pragma unroll
    for (int kc = 0; kc < 4; ++kc)
      c = wmma_bf16(a[kc], load_frag(Pfin, 8, kc, nt, lane), c);
    float bb = bfin[(nt << 4) + col];
#pragma unroll
    for (int v = 0; v < 8; ++v)
      m[nt][v] = swishf(c[v] + bb) + Xb[(size_t)(v + ro) * F + (nt << 4) + col];
  }

  residual128(m, tile, Pra, bra, lane);                       // NA layer 0
  residual128(m, tile, Pra + 2 * MATFRAG, bra + 2 * F, lane); // NA layer 1

  float* Ob = out + (size_t)e0 * F;
#pragma unroll
  for (int nt = 0; nt < 8; ++nt)
#pragma unroll
    for (int v = 0; v < 8; ++v)
      Ob[(size_t)(v + ro) * F + (nt << 4) + col] = m[nt][v];
}

extern "C" void kernel_launch(void* const* d_in, const int* in_sizes, int n_in,
                              void* d_out, int out_size, void* d_ws, size_t ws_size,
                              hipStream_t stream) {
  const float* m_input = (const float*)d_in[0];
  const float* rbf     = (const float*)d_in[1];
  const float* sbf     = (const float*)d_in[2];
  const int*   red     = (const int*)d_in[3];
  const int*   expk    = (const int*)d_in[4];
  const float* W_kj    = (const float*)d_in[5];
  const float* b_kj    = (const float*)d_in[6];
  const float* W_rbf1  = (const float*)d_in[7];
  const float* W_rbf2  = (const float*)d_in[8];
  const float* W_sbf1  = (const float*)d_in[9];
  const float* W_sbf2  = (const float*)d_in[10];
  const float* W_down  = (const float*)d_in[11];
  const float* W_up    = (const float*)d_in[12];
  const float* W_ji    = (const float*)d_in[13];
  const float* b_ji    = (const float*)d_in[14];
  const float* W_rb    = (const float*)d_in[15];
  const float* b_rb    = (const float*)d_in[16];
  const float* W_fin   = (const float*)d_in[17];
  const float* b_fin   = (const float*)d_in[18];
  const float* W_ra    = (const float*)d_in[19];
  const float* b_ra    = (const float*)d_in[20];

  const int E = in_sizes[0] / F;  // 262144 (multiple of 128)
  const int T = in_sizes[3];      // 2097152

  // d_ws layout: packed weights | m_ang bf16 [E,64] | agg fp32 [E,64]
  bf16* ws = (bf16*)d_ws;
  const size_t MAT = (size_t)MATFRAG;  // 16384 bf16 per 128x128
  const size_t HAL = MAT / 2;          // per 128x64 / 64x128
  bf16* Pkj  = ws;
  bf16* Pdn  = Pkj + MAT;
  bf16* Pup  = Pdn + HAL;
  bf16* Pji  = Pup + HAL;
  bf16* Prb  = Pji + MAT;
  bf16* Pfin = Prb + 2 * MAT;
  bf16* Pra  = Pfin + MAT;
  bf16* mang = Pra + 4 * MAT;
  float* agg = (float*)(mang + (size_t)E * ADIM);

  auto packs = [&](const float* W, bf16* P, int K, int N) {
    int total = (K / 32) * (N / 16) * 32;
    k_pack<<<(total + 255) / 256, 256, 0, stream>>>(W, P, K, N);
  };
  packs(W_kj, Pkj, F, F);
  packs(W_down, Pdn, F, ADIM);
  packs(W_up, Pup, ADIM, F);
  packs(W_ji, Pji, F, F);
  packs(W_rb, Prb, F, F);
  packs(W_rb + F * F, Prb + MAT, F, F);
  packs(W_fin, Pfin, F, F);
  for (int i = 0; i < 4; ++i)
    packs(W_ra + (size_t)i * F * F, Pra + (size_t)i * MAT, F, F);

  size_t aggN = (size_t)E * ADIM;
  k_zero<<<(int)((aggN + 255) / 256), 256, 0, stream>>>(agg, aggN);

  k_edge_pre<<<E / 128, 256, 0, stream>>>(m_input, rbf, W_rbf1, W_rbf2, b_kj, Pkj,
                                          Pdn, mang);
  k_triplet<<<(T + 255) / 256, 256, 0, stream>>>(sbf, red, expk, W_sbf1, W_sbf2,
                                                 mang, agg, T);
  k_edge_post<<<E / 128, 256, 0, stream>>>(m_input, agg, Pji, b_ji, Pup, Prb, b_rb,
                                           Pfin, b_fin, Pra, b_ra, (float*)d_out);
}
